// CrossAttention_57698590654672
// MI455X (gfx1250) — compile-verified, matrix operand order checked by reading, not run on hardware
//
#include <hip/hip_runtime.h>
#include <hip/hip_bf16.h>

typedef __attribute__((ext_vector_type(16))) __bf16 v16bf;
typedef __attribute__((ext_vector_type(8)))  __bf16 v8bf;
typedef __attribute__((ext_vector_type(8)))  float  v8f;

#define LDS_STRIDE 40   // 32 + 8 pad (bf16 elems) -> conflict-free ds_load_b128

#define OUT_BF16        0
#define OUT_BF16_TRANS  1
#define OUT_F32         2

// ---------------------------------------------------------------------------
// CDNA5 async global->LDS copy (ASYNCcnt-tracked), 16 bytes per lane.
// Inline asm: portable across toolchains whose builtin signatures differ.
// ---------------------------------------------------------------------------
__device__ __forceinline__ void async_copy_b128(void* lds_dst, const void* g_src) {
  // Low 32 bits of a generic LDS address are the LDS byte offset (ISA 10.2).
  unsigned lds_off = (unsigned)(unsigned long long)(size_t)lds_dst;
  unsigned long long ga = (unsigned long long)(size_t)g_src;
  asm volatile("global_load_async_to_lds_b128 %0, %1, off"
               :: "v"(lds_off), "v"(ga) : "memory");
}

__device__ __forceinline__ void wait_async0() {
  asm volatile("s_wait_asynccnt 0" ::: "memory");
}

// ---------------------------------------------------------------------------
// f32 -> bf16 elementwise convert
// ---------------------------------------------------------------------------
__global__ __launch_bounds__(256)
void cvt_f32_bf16(const float* __restrict__ in, __bf16* __restrict__ out, long n) {
  long i = (long)blockIdx.x * 256 + threadIdx.x;
  if (i < n) out[i] = (__bf16)in[i];
}

// ---------------------------------------------------------------------------
// Batched GEMM: Out[b] = alpha * (A[b] x Bm[b]^T) + bias
//   A : [M x K] row-major bf16 (K contiguous)
//   Bm: [N x K] row-major bf16 (nn.Linear weight layout / K^T / V^T layout)
// Block: 256 threads (8 waves). Block tile 128x128, K-step 32.
// Waves 4x2: wave rows (w&3)*32, cols (w>>2)*64 -> 2x4 WMMA tiles per wave.
// Double-buffered LDS fed by global_load_async_to_lds_b128; one barrier/step.
// ---------------------------------------------------------------------------
template <int OUT_MODE>
__global__ __launch_bounds__(256)
void gemm_bf16_wmma(const __bf16* __restrict__ A, long strideA,
                    const __bf16* __restrict__ Bm, long strideB,
                    const float* __restrict__ bias,
                    void* __restrict__ Out, long strideO,
                    int K, int lda, int ldb, int ldo, float alpha)
{
  __shared__ __bf16 sA[2][128 * LDS_STRIDE];
  __shared__ __bf16 sB[2][128 * LDS_STRIDE];

  const int bz = blockIdx.z;
  A  += (long)bz * strideA;
  Bm += (long)bz * strideB;

  const int m0 = blockIdx.y * 128;
  const int n0 = blockIdx.x * 128;

  const int t    = threadIdx.x;
  const int wave = t >> 5;
  const int lane = t & 31;
  const int wm = (wave & 3) * 32;   // wave row offset inside block tile
  const int wn = (wave >> 2) * 64;  // wave col offset inside block tile

  // WMMA 16-bit A/B fragment lane mapping (ISA 7.12.2):
  // lane<16 : row lane,    K = 0..7 and 16..23
  // lane>=16: row lane-16, K = 8..15 and 24..31
  const int lrow = lane & 15;
  const int kb   = (lane >> 4) * 8;

  v8f c[2][4] = {};

  // Stage one 128x32 A tile + 128x32 B tile into LDS buffer `buf` for k=kk.
  // 512 b128 transfers per tile; 256 threads x 2 each, per tile.
  auto stage = [&](int buf, int kk) {
#pragma unroll
    for (int r = 0; r < 2; ++r) {
      const int idx = t + r * 256;
      const int row = idx >> 2;
      const int q   = (idx & 3) * 8;
      async_copy_b128(&sA[buf][row * LDS_STRIDE + q],
                      A + (long)(m0 + row) * lda + kk + q);
      async_copy_b128(&sB[buf][row * LDS_STRIDE + q],
                      Bm + (long)(n0 + row) * ldb + kk + q);
    }
  };

  stage(0, 0);
  int cur = 0;

  for (int k0 = 0; k0 < K; k0 += 32) {
    wait_async0();       // my async copies into buf[cur] are done
    __syncthreads();     // everyone's are; also: everyone done reading buf[cur^1]

    if (k0 + 32 < K) stage(cur ^ 1, k0 + 32);

    v16bf af[2], bfr[4];
#pragma unroll
    for (int i = 0; i < 2; ++i) {
      const __bf16* pa = &sA[cur][(wm + i * 16 + lrow) * LDS_STRIDE + kb];
      v8bf alo = *(const v8bf*)pa;
      v8bf ahi = *(const v8bf*)(pa + 16);
      af[i] = __builtin_shufflevector(alo, ahi, 0,1,2,3,4,5,6,7,8,9,10,11,12,13,14,15);
    }
#pragma unroll
    for (int j = 0; j < 4; ++j) {
      const __bf16* pb = &sB[cur][(wn + j * 16 + lrow) * LDS_STRIDE + kb];
      v8bf blo = *(const v8bf*)pb;
      v8bf bhi = *(const v8bf*)(pb + 16);
      bfr[j] = __builtin_shufflevector(blo, bhi, 0,1,2,3,4,5,6,7,8,9,10,11,12,13,14,15);
    }

#pragma unroll
    for (int i = 0; i < 2; ++i)
#pragma unroll
      for (int j = 0; j < 4; ++j)
        c[i][j] = __builtin_amdgcn_wmma_f32_16x16x32_bf16(
            false, af[i], false, bfr[j], (short)0, c[i][j], false, false);

    cur ^= 1;
  }

  // Epilogue. C/D layout: VGPR r -> M = 8*(lane>=16)+r, N = lane&15.
  const int row_off = (lane >> 4) * 8;
  const int col     = lane & 15;
#pragma unroll
  for (int j = 0; j < 4; ++j) {
    const int nn = n0 + wn + j * 16 + col;
    const float bv = bias ? bias[nn] : 0.0f;
#pragma unroll
    for (int i = 0; i < 2; ++i) {
#pragma unroll
      for (int r = 0; r < 8; ++r) {
        const int mm = m0 + wm + i * 16 + row_off + r;
        const float val = c[i][j][r] * alpha + bv;
        if (OUT_MODE == OUT_BF16) {
          ((__bf16*)Out)[(long)bz * strideO + (long)mm * ldo + nn] = (__bf16)val;
        } else if (OUT_MODE == OUT_BF16_TRANS) {
          ((__bf16*)Out)[(long)bz * strideO + (long)nn * ldo + mm] = (__bf16)val;
        } else {
          ((float*)Out)[(long)bz * strideO + (long)mm * ldo + nn] = val;
        }
      }
    }
  }
}

// ---------------------------------------------------------------------------
// Row softmax: one 256-thread block per row of `ncols` f32, bf16 out.
// ---------------------------------------------------------------------------
__global__ __launch_bounds__(256)
void softmax_rows(const float* __restrict__ S, __bf16* __restrict__ P, int ncols) {
  __shared__ float red[256];
  const long row = blockIdx.x;
  const float* s = S + row * ncols;
  __bf16* p = P + row * ncols;
  const int t = threadIdx.x;

  float m = -3.0e38f;
  for (int i = t; i < ncols; i += 256) m = fmaxf(m, s[i]);
  red[t] = m;
  __syncthreads();
  for (int off = 128; off > 0; off >>= 1) {
    if (t < off) red[t] = fmaxf(red[t], red[t + off]);
    __syncthreads();
  }
  m = red[0];
  __syncthreads();

  float sum = 0.0f;
  for (int i = t; i < ncols; i += 256) sum += __expf(s[i] - m);
  red[t] = sum;
  __syncthreads();
  for (int off = 128; off > 0; off >>= 1) {
    if (t < off) red[t] += red[t + off];
    __syncthreads();
  }
  const float inv = 1.0f / red[0];

  for (int i = t; i < ncols; i += 256) p[i] = (__bf16)(__expf(s[i] - m) * inv);
}

// ---------------------------------------------------------------------------
extern "C" void kernel_launch(void* const* d_in, const int* in_sizes, int n_in,
                              void* d_out, int out_size, void* d_ws, size_t ws_size,
                              hipStream_t stream) {
  const float* x   = (const float*)d_in[0];
  const float* ctx = (const float*)d_in[1];
  const float* Wq  = (const float*)d_in[2];
  const float* bq  = (const float*)d_in[3];
  const float* Wk  = (const float*)d_in[4];
  const float* bk  = (const float*)d_in[5];
  const float* Wv  = (const float*)d_in[6];
  const float* bv  = (const float*)d_in[7];
  const float* Wo  = (const float*)d_in[8];
  const float* bo  = (const float*)d_in[9];
  float* out = (float*)d_out;

  const int B = 8, Sq = 2048, Skv = 2048, D = 1024, C = 768;
  const long MQ = (long)B * Sq;   // 16384

  char* ws = (char*)d_ws;
  size_t off = 0;
  auto alloc = [&](size_t bytes) -> char* {
    char* p = ws + off;
    off += (bytes + 255) & ~(size_t)255;
    return p;
  };
  __bf16* xb   = (__bf16*)alloc(MQ * D * 2);            // x bf16
  __bf16* cb   = (__bf16*)alloc(MQ * C * 2);            // context bf16
  __bf16* Wqb  = (__bf16*)alloc((size_t)D * D * 2);
  __bf16* Wkb  = (__bf16*)alloc((size_t)D * C * 2);
  __bf16* Wvb  = (__bf16*)alloc((size_t)D * C * 2);
  __bf16* Wob  = (__bf16*)alloc((size_t)D * D * 2);
  __bf16* Qb   = (__bf16*)alloc(MQ * D * 2);            // [B*Sq, D]
  __bf16* Kb   = (__bf16*)alloc(MQ * D * 2);            // [B*Skv, D]
  __bf16* Vt   = (__bf16*)alloc(MQ * D * 2);            // [B, D, Skv]
  float*  Sf   = (float*)alloc((size_t)B * Sq * Skv * 4); // scores f32
  __bf16* Pb   = (__bf16*)alloc((size_t)B * Sq * Skv * 2);
  __bf16* Ob   = (__bf16*)alloc(MQ * D * 2);            // attn out

  auto conv = [&](const float* in, __bf16* o, long n) {
    cvt_f32_bf16<<<dim3((unsigned)((n + 255) / 256)), dim3(256), 0, stream>>>(in, o, n);
  };
  conv(x,   xb,  MQ * D);
  conv(ctx, cb,  MQ * C);
  conv(Wq,  Wqb, (long)D * D);
  conv(Wk,  Wkb, (long)D * C);
  conv(Wv,  Wvb, (long)D * C);
  conv(Wo,  Wob, (long)D * D);

  // Q = x . Wq^T + bq            [16384 x 1024], K=1024
  gemm_bf16_wmma<OUT_BF16><<<dim3(D / 128, MQ / 128, 1), dim3(256), 0, stream>>>(
      xb, 0, Wqb, 0, bq, Qb, 0, D, D, D, D, 1.0f);

  // K = ctx . Wk^T + bk          [16384 x 1024], K=768
  gemm_bf16_wmma<OUT_BF16><<<dim3(D / 128, MQ / 128, 1), dim3(256), 0, stream>>>(
      cb, 0, Wkb, 0, bk, Kb, 0, C, C, C, D, 1.0f);

  // V^T[b] = (ctx[b] . Wv^T + bv)^T   batched, transposed store [B, D, Skv]
  gemm_bf16_wmma<OUT_BF16_TRANS><<<dim3(D / 128, Skv / 128, B), dim3(256), 0, stream>>>(
      cb, (long)Skv * C, Wvb, 0, bv, Vt, (long)D * Skv, C, C, C, Skv, 1.0f);

  // S[b] = Q[b] . K[b]^T * (1/sqrt(D))   [B, Sq, Skv] f32
  gemm_bf16_wmma<OUT_F32><<<dim3(Skv / 128, Sq / 128, B), dim3(256), 0, stream>>>(
      Qb, (long)Sq * D, Kb, (long)Skv * D, nullptr,
      Sf, (long)Sq * Skv, D, D, D, Skv, 0.03125f /* 1/sqrt(1024) */);

  // P = softmax_rows(S) -> bf16
  softmax_rows<<<dim3((unsigned)(B * Sq)), dim3(256), 0, stream>>>(Sf, Pb, Skv);

  // O[b] = P[b] . V[b]   via V^T as [N=D rows, K=Skv]   -> [B*Sq, D] bf16
  gemm_bf16_wmma<OUT_BF16><<<dim3(D / 128, Sq / 128, B), dim3(256), 0, stream>>>(
      Pb, (long)Sq * Skv, Vt, (long)D * Skv, nullptr,
      Ob, (long)Sq * D, Skv, Skv, Skv, D, 1.0f);

  // out = O . Wo^T + bo   [16384 x 1024] f32
  gemm_bf16_wmma<OUT_F32><<<dim3(D / 128, MQ / 128, 1), dim3(256), 0, stream>>>(
      Ob, 0, Wob, 0, bo, out, 0, D, D, D, D, 1.0f);
}
